// QuantMultiheadAttention_79869211836976
// MI455X (gfx1250) — matrix operand order repaired and checked by reading, not compiled, not run on hardware
//
#include <hip/hip_runtime.h>
#include <hip/hip_bf16.h>

#define SEQ    2048
#define BSZ    4
#define EMBED  1024
#define NHEADS 16
#define HDIM   64
#define TOKENS (SEQ * BSZ)   // 8192

typedef __attribute__((ext_vector_type(16))) _Float16 v16h;
typedef __attribute__((ext_vector_type(8)))  _Float16 v8h;
typedef __attribute__((ext_vector_type(4)))  _Float16 v4h;
typedef __attribute__((ext_vector_type(8)))  float    v8f;

// ---------------------------------------------------------------------------
// Fragment loaders (CDNA5 wave32 WMMA VGPR layouts, CDNA5 ISA)
// A 16x32 f16: lane L holds row m=L%16; v0..v3 = K[(L/16)*8 .. +7],
//              v4..v7 = K[16+(L/16)*8 .. +7]  -> two 16B contiguous loads.
// B 32x16 f16 (stored as B^T row-major, K contiguous): lane L holds col n=L%16,
//              v0..v3 = K[(L/16)*8 .. +7], v4..v7 = K[16+(L/16)*8 .. +7] -> two 16B loads.
// ---------------------------------------------------------------------------
__device__ __forceinline__ v16h load_a_f16(const _Float16* __restrict__ p) {
  v8h lo = *(const v8h*)p;
  v8h hi = *(const v8h*)(p + 16);
  v16h r;
#pragma unroll
  for (int i = 0; i < 8; ++i) { r[i] = lo[i]; r[i + 8] = hi[i]; }
  return r;
}

__device__ __forceinline__ v16h load_b_f16(const _Float16* __restrict__ p) {
  v8h lo = *(const v8h*)p;
  v8h hi = *(const v8h*)(p + 16);
  v16h r;
#pragma unroll
  for (int i = 0; i < 8; ++i) { r[i] = lo[i]; r[i + 8] = hi[i]; }
  return r;
}

__device__ __forceinline__ v16h load_a_f32(const float* __restrict__ p) {
  // fp32 source, convert to f16 in-register (attn matrix read path)
  const float4* p4 = (const float4*)p;
  float4 a0 = p4[0], a1 = p4[1];   // K+0..7
  float4 b0 = p4[4], b1 = p4[5];   // K+16..23
  v16h r;
  r[0]  = (_Float16)a0.x; r[1]  = (_Float16)a0.y; r[2]  = (_Float16)a0.z; r[3]  = (_Float16)a0.w;
  r[4]  = (_Float16)a1.x; r[5]  = (_Float16)a1.y; r[6]  = (_Float16)a1.z; r[7]  = (_Float16)a1.w;
  r[8]  = (_Float16)b0.x; r[9]  = (_Float16)b0.y; r[10] = (_Float16)b0.z; r[11] = (_Float16)b0.w;
  r[12] = (_Float16)b1.x; r[13] = (_Float16)b1.y; r[14] = (_Float16)b1.z; r[15] = (_Float16)b1.w;
  return r;
}

// ---------------------------------------------------------------------------
// fp32 -> fp16 conversion, 4 elements / thread
// ---------------------------------------------------------------------------
__global__ void __launch_bounds__(256) cvt_f32_f16_kernel(
    const float* __restrict__ src, _Float16* __restrict__ dst, size_t n4) {
  size_t i = (size_t)blockIdx.x * blockDim.x + threadIdx.x;
  if (i < n4) {
    float4 v = ((const float4*)src)[i];
    v4h o;
    o[0] = (_Float16)v.x; o[1] = (_Float16)v.y;
    o[2] = (_Float16)v.z; o[3] = (_Float16)v.w;
    ((v4h*)dst)[i] = o;
  }
}

// ---------------------------------------------------------------------------
// QKV projection: C[t,f] = sum_e x16[t,e] * W_in16[f,e] + b_in[f]
// Block tile 128x128, 8 waves (2x4), wave tile 64x32 = 4x2 WMMA accumulators.
// Epilogue scatters to q [BH,S,D] (pre-scaled by 1/sqrt(D)), k [BH,S,D],
// and vT [BH,D,S].
// ---------------------------------------------------------------------------
__global__ void __launch_bounds__(256) qkv_wmma_kernel(
    const _Float16* __restrict__ A,   // x16   [TOKENS, EMBED]
    const _Float16* __restrict__ Bw,  // Wi16  [3*EMBED, EMBED]
    const float*    __restrict__ bias,// b_in  [3*EMBED]
    _Float16* __restrict__ q16,
    _Float16* __restrict__ k16,
    _Float16* __restrict__ v16t) {
  const int lane = threadIdx.x & 31;
  const int wave = threadIdx.x >> 5;
  const int row0 = blockIdx.x * 128 + (wave >> 2) * 64;
  const int col0 = blockIdx.y * 128 + (wave & 3) * 32;
  const int m_l  = lane & 15;
  const int kq_a = (lane >> 4) * 8;
  const int kq_b = (lane >> 4) * 8;

  v8f acc[4][2] = {};

  for (int k0 = 0; k0 < EMBED; k0 += 32) {
    __builtin_prefetch((const void*)(A + (size_t)(row0 + m_l) * EMBED + k0 + 128), 0, 0);
    v16h af[4];
#pragma unroll
    for (int tm = 0; tm < 4; ++tm)
      af[tm] = load_a_f16(A + (size_t)(row0 + tm * 16 + m_l) * EMBED + k0 + kq_a);
    v16h bf[2];
#pragma unroll
    for (int tn = 0; tn < 2; ++tn)
      bf[tn] = load_b_f16(Bw + (size_t)(col0 + tn * 16 + m_l) * EMBED + k0 + kq_b);
#pragma unroll
    for (int tm = 0; tm < 4; ++tm)
#pragma unroll
      for (int tn = 0; tn < 2; ++tn)
        acc[tm][tn] = __builtin_amdgcn_wmma_f32_16x16x32_f16(
            false, af[tm], false, bf[tn], (short)0, acc[tm][tn], false, false);
  }

  const float scale = 0.125f;  // 1/sqrt(64)
#pragma unroll
  for (int tm = 0; tm < 4; ++tm) {
#pragma unroll
    for (int tn = 0; tn < 2; ++tn) {
      const int f = col0 + tn * 16 + (lane & 15);
      const float bv = bias[f];
      const int chunk = f >> 10;        // /EMBED
      const int r     = f & 1023;
      const int h     = r >> 6;         // /HDIM
      const int d     = r & 63;
#pragma unroll
      for (int i = 0; i < 8; ++i) {
        const int row = row0 + tm * 16 + (lane >> 4) * 8 + i;  // token index
        const int s   = row >> 2;       // /BSZ
        const int b   = row & 3;
        const int bh  = b * NHEADS + h;
        const float c = acc[tm][tn][i] + bv;
        if (chunk == 0) {
          q16[((size_t)bh * SEQ + s) * HDIM + d] = (_Float16)(c * scale);
        } else if (chunk == 1) {
          k16[((size_t)bh * SEQ + s) * HDIM + d] = (_Float16)c;
        } else {
          v16t[((size_t)bh * HDIM + d) * SEQ + s] = (_Float16)c;
        }
      }
    }
  }
}

// ---------------------------------------------------------------------------
// Fused scores + softmax, one (b,h) slice per blockIdx.y.
// Block owns 32 full rows of the 2048-wide score matrix, held in a 256KB LDS
// strip (fits CDNA5's 320KB-per-WGP LDS).  8 waves each compute a 32x256
// column band with WMMA (q fragments loaded once, reused for all 8 column
// chunks), spill fp32 scores to LDS, then the block does an in-LDS row
// softmax and writes the *normalized* attn to global exactly once.
// This removes the standalone softmax pass (saves ~2.1 GB of HBM traffic).
// ---------------------------------------------------------------------------
__global__ void __launch_bounds__(256) scores_softmax_wmma_kernel(
    const _Float16* __restrict__ q16,
    const _Float16* __restrict__ k16,
    float* __restrict__ attn) {
  extern __shared__ float smem[];
  float* tile   = smem;                 // [32][2048] fp32 scores
  float* rowinv = smem + 32 * SEQ;      // [32]
  float* red    = rowinv + 32;          // [32][8] reduction partials

  const int bh = blockIdx.y;
  const _Float16* Aq = q16 + (size_t)bh * SEQ * HDIM;
  const _Float16* Bk = k16 + (size_t)bh * SEQ * HDIM;
  const int r0 = blockIdx.x * 32;
  float* Crow = attn + ((size_t)bh * SEQ + r0) * SEQ;

  const int lane = threadIdx.x & 31;
  const int wave = threadIdx.x >> 5;
  const int m_l  = lane & 15;
  const int kq_a = (lane >> 4) * 8;
  const int kq_b = (lane >> 4) * 8;

  // q fragments: 2 row-tiles x 2 k-steps, loaded once (scale already folded in)
  v16h af[2][2];
#pragma unroll
  for (int tm = 0; tm < 2; ++tm)
#pragma unroll
    for (int ks = 0; ks < 2; ++ks)
      af[tm][ks] = load_a_f16(Aq + (size_t)(r0 + tm * 16 + m_l) * HDIM + ks * 32 + kq_a);

  for (int cc = 0; cc < 8; ++cc) {
    const int col0 = wave * 256 + cc * 32;
    v8f acc[2][2] = {};
#pragma unroll
    for (int ks = 0; ks < 2; ++ks) {
      v16h bf[2];
#pragma unroll
      for (int tn = 0; tn < 2; ++tn)
        bf[tn] = load_b_f16(Bk + (size_t)(col0 + tn * 16 + m_l) * HDIM + ks * 32 + kq_b);
#pragma unroll
      for (int tm = 0; tm < 2; ++tm)
#pragma unroll
        for (int tn = 0; tn < 2; ++tn)
          acc[tm][tn] = __builtin_amdgcn_wmma_f32_16x16x32_f16(
              false, af[tm][ks], false, bf[tn], (short)0, acc[tm][tn], false, false);
    }
#pragma unroll
    for (int tm = 0; tm < 2; ++tm)
#pragma unroll
      for (int tn = 0; tn < 2; ++tn) {
        const int cl = col0 + tn * 16 + (lane & 15);
#pragma unroll
        for (int i = 0; i < 8; ++i) {
          const int rl = tm * 16 + (lane >> 4) * 8 + i;
          tile[rl * SEQ + cl] = acc[tm][tn][i];
        }
      }
  }
  __syncthreads();

  // ---- in-LDS softmax over the 32 rows: 8 threads per row --------------
  const int srow = threadIdx.x >> 3;
  const int ssub = threadIdx.x & 7;
  float* trow = tile + srow * SEQ + ssub * 256;

  float m = -3.0e38f;
  for (int j = 0; j < 256; ++j) m = fmaxf(m, trow[j]);
  red[srow * 8 + ssub] = m;
  __syncthreads();
  float rm = red[srow * 8];
#pragma unroll
  for (int j = 1; j < 8; ++j) rm = fmaxf(rm, red[srow * 8 + j]);
  __syncthreads();

  float sum = 0.0f;
  for (int j = 0; j < 256; ++j) { float e = __expf(trow[j] - rm); trow[j] = e; sum += e; }
  red[srow * 8 + ssub] = sum;
  __syncthreads();
  if (ssub == 0) {
    float tot = 0.0f;
#pragma unroll
    for (int j = 0; j < 8; ++j) tot += red[srow * 8 + j];
    rowinv[srow] = 1.0f / tot;
  }
  __syncthreads();

  // ---- coalesced normalized write-out (float4) -------------------------
  const float4* t4 = (const float4*)tile;
  float4* c4 = (float4*)Crow;
  for (int r = 0; r < 32; ++r) {
    const float inv = rowinv[r];
#pragma unroll
    for (int j = 0; j < 2; ++j) {
      const int idx = r * (SEQ / 4) + threadIdx.x + 256 * j;
      float4 v = t4[idx];
      v.x *= inv; v.y *= inv; v.z *= inv; v.w *= inv;
      c4[idx] = v;
    }
  }
}

// ---------------------------------------------------------------------------
// AV: per (b,h): head[s,d] = sum_t attn[s,t] * v[t,d]
// A = attn fp32 (converted to f16 in-register), B = vT [D, S] f16.
// Block tile 256x64, 8 waves along M, wave tile 32x64 = 2x4 accumulators.
// Output fp16 head in [S, B, E] layout (E = h*64+d).
// ---------------------------------------------------------------------------
__global__ void __launch_bounds__(256) av_wmma_kernel(
    const float*    __restrict__ attn,
    const _Float16* __restrict__ v16t,
    _Float16* __restrict__ head16) {
  const int bh = blockIdx.y;
  const float*    Aa = attn + (size_t)bh * SEQ * SEQ;
  const _Float16* Bv = v16t + (size_t)bh * HDIM * SEQ;
  const int b = bh >> 4;    // bh = b*NHEADS + h
  const int h = bh & 15;

  const int lane = threadIdx.x & 31;
  const int wave = threadIdx.x >> 5;
  const int row0 = blockIdx.x * 256 + wave * 32;
  const int m_l  = lane & 15;
  const int kq_a = (lane >> 4) * 8;
  const int kq_b = (lane >> 4) * 8;

  v8f acc[2][4] = {};

  for (int k0 = 0; k0 < SEQ; k0 += 32) {
    __builtin_prefetch((const void*)(Aa + (size_t)(row0 + m_l) * SEQ + k0 + 128), 0, 0);
    v16h af[2];
#pragma unroll
    for (int tm = 0; tm < 2; ++tm)
      af[tm] = load_a_f32(Aa + (size_t)(row0 + tm * 16 + m_l) * SEQ + k0 + kq_a);
    v16h bf[4];
#pragma unroll
    for (int tn = 0; tn < 4; ++tn)
      bf[tn] = load_b_f16(Bv + (size_t)(tn * 16 + m_l) * SEQ + k0 + kq_b);
#pragma unroll
    for (int tm = 0; tm < 2; ++tm)
#pragma unroll
      for (int tn = 0; tn < 4; ++tn)
        acc[tm][tn] = __builtin_amdgcn_wmma_f32_16x16x32_f16(
            false, af[tm], false, bf[tn], (short)0, acc[tm][tn], false, false);
  }

#pragma unroll
  for (int tm = 0; tm < 2; ++tm)
#pragma unroll
    for (int tn = 0; tn < 4; ++tn) {
      const int d = tn * 16 + (lane & 15);
#pragma unroll
      for (int i = 0; i < 8; ++i) {
        const int s = row0 + tm * 16 + (lane >> 4) * 8 + i;
        head16[(((size_t)s * BSZ + b) * NHEADS + h) * HDIM + d] = (_Float16)acc[tm][tn][i];
      }
    }
}

// ---------------------------------------------------------------------------
// Output projection: out[t,f] = sum_e head16[t,e] * W_out16[f,e] + b_out[f]
// ---------------------------------------------------------------------------
__global__ void __launch_bounds__(256) outproj_wmma_kernel(
    const _Float16* __restrict__ A,   // head16 [TOKENS, EMBED]
    const _Float16* __restrict__ Bw,  // Wo16   [EMBED, EMBED]
    const float*    __restrict__ bias,
    float* __restrict__ out) {
  const int lane = threadIdx.x & 31;
  const int wave = threadIdx.x >> 5;
  const int row0 = blockIdx.x * 128 + (wave >> 2) * 64;
  const int col0 = blockIdx.y * 128 + (wave & 3) * 32;
  const int m_l  = lane & 15;
  const int kq_a = (lane >> 4) * 8;
  const int kq_b = (lane >> 4) * 8;

  v8f acc[4][2] = {};

  for (int k0 = 0; k0 < EMBED; k0 += 32) {
    __builtin_prefetch((const void*)(A + (size_t)(row0 + m_l) * EMBED + k0 + 128), 0, 0);
    v16h af[4];
#pragma unroll
    for (int tm = 0; tm < 4; ++tm)
      af[tm] = load_a_f16(A + (size_t)(row0 + tm * 16 + m_l) * EMBED + k0 + kq_a);
    v16h bf[2];
#pragma unroll
    for (int tn = 0; tn < 2; ++tn)
      bf[tn] = load_b_f16(Bw + (size_t)(col0 + tn * 16 + m_l) * EMBED + k0 + kq_b);
#pragma unroll
    for (int tm = 0; tm < 4; ++tm)
#pragma unroll
      for (int tn = 0; tn < 2; ++tn)
        acc[tm][tn] = __builtin_amdgcn_wmma_f32_16x16x32_f16(
            false, af[tm], false, bf[tn], (short)0, acc[tm][tn], false, false);
  }

#pragma unroll
  for (int tm = 0; tm < 4; ++tm)
#pragma unroll
    for (int tn = 0; tn < 2; ++tn) {
      const int col = col0 + tn * 16 + (lane & 15);
      const float bv = bias[col];
#pragma unroll
      for (int i = 0; i < 8; ++i) {
        const int row = row0 + tm * 16 + (lane >> 4) * 8 + i;
        out[(size_t)row * EMBED + col] = acc[tm][tn][i] + bv;
      }
    }
}

// ---------------------------------------------------------------------------
// Host launch
// ---------------------------------------------------------------------------
extern "C" void kernel_launch(void* const* d_in, const int* in_sizes, int n_in,
                              void* d_out, int out_size, void* d_ws, size_t ws_size,
                              hipStream_t stream) {
  const float* x     = (const float*)d_in[0];   // [S, B, E]
  const float* W_in  = (const float*)d_in[1];   // [3E, E]
  const float* b_in  = (const float*)d_in[2];   // [3E]
  const float* W_out = (const float*)d_in[3];   // [E, E]
  const float* b_out = (const float*)d_in[4];   // [E]

  float* out  = (float*)d_out;                            // [S*B*E]
  float* attn = out + (size_t)SEQ * BSZ * EMBED;          // [B*H*S*S]

  char* ws = (char*)d_ws;
  size_t off = 0;
  auto take = [&](size_t elems) -> _Float16* {
    _Float16* p = (_Float16*)(ws + off);
    off += ((elems * sizeof(_Float16)) + 255) & ~(size_t)255;
    return p;
  };
  _Float16* x16  = take((size_t)TOKENS * EMBED);
  _Float16* wi16 = take((size_t)3 * EMBED * EMBED);
  _Float16* wo16 = take((size_t)EMBED * EMBED);
  _Float16* q16  = take((size_t)BSZ * NHEADS * SEQ * HDIM);
  _Float16* k16  = take((size_t)BSZ * NHEADS * SEQ * HDIM);
  _Float16* v16t = take((size_t)BSZ * NHEADS * SEQ * HDIM);
  _Float16* h16  = take((size_t)TOKENS * EMBED);
  (void)ws_size; (void)in_sizes; (void)n_in; (void)out_size;

  auto cvt = [&](const float* s, _Float16* d, size_t n) {
    size_t n4 = n / 4;
    cvt_f32_f16_kernel<<<dim3((unsigned)((n4 + 255) / 256)), dim3(256), 0, stream>>>(s, d, n4);
  };
  cvt(x,     x16,  (size_t)TOKENS * EMBED);
  cvt(W_in,  wi16, (size_t)3 * EMBED * EMBED);
  cvt(W_out, wo16, (size_t)EMBED * EMBED);

  qkv_wmma_kernel<<<dim3(TOKENS / 128, (3 * EMBED) / 128), dim3(256), 0, stream>>>(
      x16, wi16, b_in, q16, k16, v16t);

  // 256KB score strip + reduction scratch in dynamic LDS (CDNA5: 320KB/WGP)
  const size_t smem_bytes = (size_t)(32 * SEQ + 32 + 32 * 8) * sizeof(float);
  scores_softmax_wmma_kernel<<<dim3(SEQ / 32, BSZ * NHEADS), dim3(256), smem_bytes, stream>>>(
      q16, k16, attn);

  av_wmma_kernel<<<dim3(SEQ / 256, BSZ * NHEADS), dim3(256), 0, stream>>>(
      attn, v16t, h16);

  outproj_wmma_kernel<<<dim3(TOKENS / 128, EMBED / 128), dim3(256), 0, stream>>>(
      h16, wo16, b_out, out);
}
